// CustomLSTM_32882269618463
// MI455X (gfx1250) — matrix operand order repaired and checked by reading, not compiled, not run on hardware
//
#include <hip/hip_runtime.h>
#include <hip/hip_bf16.h>
#include <math.h>
#include <stdint.h>

#define B_ 64
#define T_ 512
#define I_ 1024
#define H_ 1024
#define NG 4096   // 4 * H
#define KC 32     // K per WMMA step (bf16 16x16x32)

#define ASYNC_LDS 1   // use gfx1250 GLOBAL_LOAD_ASYNC_TO_LDS_B128 for tile staging

typedef __bf16 bf16_t;
typedef __attribute__((ext_vector_type(16))) __bf16 v16bf;
typedef __attribute__((ext_vector_type(8)))  __bf16 v8bf;
typedef __attribute__((ext_vector_type(8)))  float  v8f;

__device__ __forceinline__ float sigmoidf_(float x) { return 1.0f / (1.0f + __expf(-x)); }

__device__ __forceinline__ v16bf pack16_(v8bf lo, v8bf hi) {
    v16bf r;
#pragma unroll
    for (int i = 0; i < 8; ++i) { r[i] = lo[i]; r[i + 8] = hi[i]; }
    return r;
}

// Copy 16B global -> LDS. Async variant uses the CDNA5 TDM-adjacent async path
// (vmem writes LDS directly, tracked by ASYNCcnt), avoiding the VGPR bounce.
__device__ __forceinline__ void stage_b128(bf16_t* lds_dst, const bf16_t* gsrc) {
#if ASYNC_LDS
    asm volatile("global_load_async_to_lds_b128 %0, %1, off"
                 :: "v"((uint32_t)(uintptr_t)lds_dst),   // LDS byte offset (flat addr[31:0])
                    "v"((uint64_t)(uintptr_t)gsrc)
                 : "memory");
#else
    *(v8bf*)lds_dst = *(const v8bf*)gsrc;
#endif
}

__device__ __forceinline__ void stage_wait() {
#if ASYNC_LDS
    asm volatile("s_wait_asynccnt 0x0" ::: "memory");
#endif
}

// ---------------- prep kernels ----------------

// x (B,T,I) f32 -> xbf [(t*B+b)][I] bf16   (A-matrix, row-major K contiguous)
__global__ void k_cast_x(const float* __restrict__ x, bf16_t* __restrict__ xbf) {
    size_t idx = (size_t)blockIdx.x * blockDim.x + threadIdx.x; // T*B*I
    int i = (int)(idx & (I_ - 1));
    int r = (int)(idx >> 10);
    int t = r >> 6;     // r / B
    int b = r & 63;     // r % B
    xbf[idx] = (bf16_t)x[((size_t)b * T_ + t) * I_ + i];
}

// transpose 4 gate matrices [K][N] f32 -> packed bf16 [(g*N+n)][K]  (B-operand layout)
__global__ void k_pack_gates(const float* __restrict__ m0, const float* __restrict__ m1,
                             const float* __restrict__ m2, const float* __restrict__ m3,
                             bf16_t* __restrict__ dst) {
    size_t idx = (size_t)blockIdx.x * blockDim.x + threadIdx.x; // 4*1024*1024
    int k = (int)(idx & (I_ - 1));
    size_t n4 = idx >> 10;          // g*H + h
    int g = (int)(n4 >> 10);
    int h = (int)(n4 & (H_ - 1));
    const float* m = (g == 0) ? m0 : (g == 1) ? m1 : (g == 2) ? m2 : m3;
    dst[idx] = (bf16_t)m[(size_t)k * H_ + h];
}

__global__ void k_bias(const float* __restrict__ b0, const float* __restrict__ b1,
                       const float* __restrict__ b2, const float* __restrict__ b3,
                       float* __restrict__ biasAll) {
    int idx = blockIdx.x * blockDim.x + threadIdx.x; // 4096
    int g = idx >> 10, h = idx & (H_ - 1);
    const float* b = (g == 0) ? b0 : (g == 1) ? b1 : (g == 2) ? b2 : b3;
    biasAll[idx] = b[h];
}

__global__ void k_init(bf16_t* __restrict__ h0, float* __restrict__ c0) {
    int idx = blockIdx.x * blockDim.x + threadIdx.x; // B*H
    h0[idx] = (bf16_t)0.0f;
    c0[idx] = 0.0f;
}

// ---------------- phase 1: xW = x*W + b ----------------
// C[r][n] = sum_k A[r][k]*Bt[n][k] + bias[n];  M=32768 N=4096 K=1024
__launch_bounds__(256)
__global__ void k_gemm_xw(const bf16_t* __restrict__ A,   // [32768][1024]
                          const bf16_t* __restrict__ Bt,  // [4096][1024]
                          const float*  __restrict__ bias,
                          float*        __restrict__ C) { // [32768][4096]
    __shared__ __align__(16) bf16_t lA[128 * 40];
    __shared__ __align__(16) bf16_t lB[128 * 40];
    const int tid  = threadIdx.x;
    const int row0 = blockIdx.x * 128;
    const int col0 = blockIdx.y * 128;
    const int w = tid >> 5, lane = tid & 31;
    const int lm = lane & 15, lk = lane >> 4;
    const int mwave = (w & 1) * 64;
    const int nwave = (w >> 1) * 32;

    // per-thread staging coordinates (two rows, 64 apart)
    const int sr = tid >> 2, sch = tid & 3;
    const bf16_t* gA0 = A  + (size_t)(row0 + sr)      * I_ + sch * 8;
    const bf16_t* gA1 = A  + (size_t)(row0 + sr + 64) * I_ + sch * 8;
    const bf16_t* gB0 = Bt + (size_t)(col0 + sr)      * I_ + sch * 8;
    const bf16_t* gB1 = Bt + (size_t)(col0 + sr + 64) * I_ + sch * 8;
    bf16_t* dA0 = &lA[sr * 40 + sch * 8];
    bf16_t* dA1 = &lA[(sr + 64) * 40 + sch * 8];
    bf16_t* dB0 = &lB[sr * 40 + sch * 8];
    bf16_t* dB1 = &lB[(sr + 64) * 40 + sch * 8];

    v8f acc[4][2];
#pragma unroll
    for (int mt = 0; mt < 4; ++mt)
#pragma unroll
        for (int nt = 0; nt < 2; ++nt) acc[mt][nt] = {};

    for (int kt = 0; kt < I_ / KC; ++kt) {
        const int k0 = kt * KC;
        __syncthreads();
        stage_b128(dA0, gA0 + k0);
        stage_b128(dA1, gA1 + k0);
        stage_b128(dB0, gB0 + k0);
        stage_b128(dB1, gB1 + k0);
        stage_wait();
        __syncthreads();

        v16bf af[4], bq[2];
#pragma unroll
        for (int mt = 0; mt < 4; ++mt) {
            int r = mwave + mt * 16 + lm;
            v8bf lo = *(const v8bf*)&lA[r * 40 + lk * 8];
            v8bf hi = *(const v8bf*)&lA[r * 40 + 16 + lk * 8];
            af[mt] = pack16_(lo, hi);
        }
#pragma unroll
        for (int nt = 0; nt < 2; ++nt) {
            int n = nwave + nt * 16 + lm;
            v8bf lo = *(const v8bf*)&lB[n * 40 + lk * 8];
            v8bf hi = *(const v8bf*)&lB[n * 40 + 16 + lk * 8];
            bq[nt] = pack16_(lo, hi);
        }
#pragma unroll
        for (int mt = 0; mt < 4; ++mt)
#pragma unroll
            for (int nt = 0; nt < 2; ++nt)
                acc[mt][nt] = __builtin_amdgcn_wmma_f32_16x16x32_bf16(
                    false, af[mt], false, bq[nt], (short)0, acc[mt][nt], false, false);
    }

#pragma unroll
    for (int mt = 0; mt < 4; ++mt)
#pragma unroll
        for (int nt = 0; nt < 2; ++nt)
#pragma unroll
            for (int v = 0; v < 8; ++v) {
                int row = row0 + mwave + mt * 16 + v + lk * 8;
                int col = col0 + nwave + nt * 16 + lm;
                C[(size_t)row * NG + col] = acc[mt][nt][v] + bias[col];
            }
}

// ---------------- phase 2: one recurrence step ----------------
// WG owns 16 h-columns across all 4 gates. 8 waves: gate = w&3, row-half = w>>2.
__launch_bounds__(256)
__global__ void k_step(int t,
                       const bf16_t* __restrict__ hin,   // [64][1024] bf16
                       bf16_t*       __restrict__ hout,  // [64][1024] bf16
                       float*        __restrict__ cbuf,  // [64][1024] f32
                       const float*  __restrict__ xW,    // [T*B][4096]
                       const bf16_t* __restrict__ Ut,    // [4096][1024]
                       float*        __restrict__ outseq) { // [B][T][H]
    __shared__ __align__(16) bf16_t lH[64 * 40];
    __shared__ float pre[4 * 64 * 16];
    const int tid = threadIdx.x;
    const int hc0 = blockIdx.x * 16;
    const int w = tid >> 5, lane = tid & 31;
    const int lm = lane & 15, lk = lane >> 4;
    const int g = w & 3, mh = (w >> 2) * 32;
    const size_t ncol = (size_t)(g * H_ + hc0 + lm);

    const int sr = tid >> 2, sch = tid & 3;                     // 64 rows x 4 chunks
    const bf16_t* gH = hin + (size_t)sr * H_ + sch * 8;
    bf16_t* dH = &lH[sr * 40 + sch * 8];

    v8f acc0 = {}, acc1 = {};
    for (int kt = 0; kt < I_ / KC; ++kt) {
        const int k0 = kt * KC;
        __syncthreads();
        stage_b128(dH, gH + k0);                                // 64x32 slice of h
        stage_wait();
        __syncthreads();

        const bf16_t* up = Ut + ncol * H_ + k0 + lk * 8;
        __builtin_prefetch(up + KC, 0, 1);
        v16bf bq = pack16_(*(const v8bf*)up, *(const v8bf*)(up + 16));

        v16bf a0, a1;
        {
            int r0 = mh + lm, r1 = mh + 16 + lm;
            a0 = pack16_(*(const v8bf*)&lH[r0 * 40 + lk * 8],
                         *(const v8bf*)&lH[r0 * 40 + 16 + lk * 8]);
            a1 = pack16_(*(const v8bf*)&lH[r1 * 40 + lk * 8],
                         *(const v8bf*)&lH[r1 * 40 + 16 + lk * 8]);
        }
        acc0 = __builtin_amdgcn_wmma_f32_16x16x32_bf16(false, a0, false, bq, (short)0, acc0, false, false);
        acc1 = __builtin_amdgcn_wmma_f32_16x16x32_bf16(false, a1, false, bq, (short)0, acc1, false, false);
    }

#pragma unroll
    for (int v = 0; v < 8; ++v) {
        pre[(g * 64 + mh +      v + lk * 8) * 16 + lm] = acc0[v];
        pre[(g * 64 + mh + 16 + v + lk * 8) * 16 + lm] = acc1[v];
    }
    __syncthreads();

    const float* xw_t = xW + (size_t)t * B_ * NG;
#pragma unroll
    for (int e = tid; e < B_ * 16; e += 256) {
        int b = e >> 4, hc = e & 15, h = hc0 + hc;
        const float* xr = xw_t + (size_t)b * NG;
        float pi = pre[(0 * 64 + b) * 16 + hc] + xr[0 * H_ + h];
        float pf = pre[(1 * 64 + b) * 16 + hc] + xr[1 * H_ + h];
        float pc = pre[(2 * 64 + b) * 16 + hc] + xr[2 * H_ + h];
        float po = pre[(3 * 64 + b) * 16 + hc] + xr[3 * H_ + h];
        float it = sigmoidf_(pi), ft = sigmoidf_(pf), gt = sigmoidf_(pc), ot = sigmoidf_(po);
        size_t ci = (size_t)b * H_ + h;
        float cn = ft + cbuf[ci] + it * gt;   // reference: c = f + c + i*g
        float hn = ot + tanhf(cn);            // reference: h = o + tanh(c)
        cbuf[ci] = cn;
        outseq[(size_t)b * T_ * H_ + (size_t)t * H_ + h] = hn;
        hout[ci] = (bf16_t)hn;
    }
}

// ---------------- final outputs ----------------
__global__ void k_final(const float* __restrict__ outseq, const float* __restrict__ cbuf,
                        float* __restrict__ out) {
    int idx = blockIdx.x * blockDim.x + threadIdx.x; // B*H
    int b = idx >> 10, h = idx & (H_ - 1);
    out[idx] = outseq[(size_t)b * T_ * H_ + (size_t)(T_ - 1) * H_ + h]; // h_t
    out[B_ * H_ + idx] = cbuf[idx];                                     // c_t
}

extern "C" void kernel_launch(void* const* d_in, const int* in_sizes, int n_in,
                              void* d_out, int out_size, void* d_ws, size_t ws_size,
                              hipStream_t stream) {
    const float* x   = (const float*)d_in[0];
    const float* w_i = (const float*)d_in[1];
    const float* u_i = (const float*)d_in[2];
    const float* b_i = (const float*)d_in[3];
    const float* w_f = (const float*)d_in[4];
    const float* u_f = (const float*)d_in[5];
    const float* b_f = (const float*)d_in[6];
    const float* w_c = (const float*)d_in[7];
    const float* u_c = (const float*)d_in[8];
    const float* b_c = (const float*)d_in[9];
    const float* w_o = (const float*)d_in[10];
    const float* u_o = (const float*)d_in[11];
    const float* b_o = (const float*)d_in[12];
    float* out = (float*)d_out;

    char* ws = (char*)d_ws;
    bf16_t* xbf     = (bf16_t*)(ws);                    // 64 MB
    bf16_t* wt      = (bf16_t*)(ws +  67108864ull);     //  8 MB
    bf16_t* ut      = (bf16_t*)(ws +  75497472ull);     //  8 MB
    float*  biasAll = (float* )(ws +  83886080ull);     // 16 KB
    float*  xW      = (float* )(ws +  83902464ull);     // 512 MB
    bf16_t* hbf0    = (bf16_t*)(ws + 620773376ull);     // 128 KB
    bf16_t* hbf1    = (bf16_t*)(ws + 620904448ull);     // 128 KB
    float*  cbuf    = (float* )(ws + 621035520ull);     // 256 KB

    k_cast_x    <<<131072, 256, 0, stream>>>(x, xbf);
    k_pack_gates<<<16384,  256, 0, stream>>>(w_i, w_f, w_c, w_o, wt);
    k_pack_gates<<<16384,  256, 0, stream>>>(u_i, u_f, u_c, u_o, ut);
    k_bias      <<<16,     256, 0, stream>>>(b_i, b_f, b_c, b_o, biasAll);
    k_init      <<<256,    256, 0, stream>>>(hbf0, cbuf);

    dim3 g1(256, 32); // M/128, N/128
    k_gemm_xw<<<g1, 256, 0, stream>>>(xbf, wt, biasAll, xW);

    float* outseq = out + 2 * B_ * H_;
    for (int t = 0; t < T_; ++t) {
        const bf16_t* hin = (t & 1) ? hbf1 : hbf0;
        bf16_t* hout      = (t & 1) ? hbf0 : hbf1;
        k_step<<<64, 256, 0, stream>>>(t, hin, hout, cbuf, xW, ut, outseq);
    }
    k_final<<<256, 256, 0, stream>>>(outseq, cbuf, out);
}
